// MambaBlock_40578851012782
// MI455X (gfx1250) — compile-verified, matrix operand order checked by reading, not run on hardware
//
#include <hip/hip_runtime.h>
#include <hip/hip_bf16.h>

// ---------------------------------------------------------------------------
// Mamba block for MI455X (gfx1250, wave32, WMMA).
// GEMMs: bf16 inputs, f32 accumulate via v_wmma_f32_16x16x32_bf16,
//        double-buffered LDS staged with global_load_async_to_lds_b128.
// Selective scan: fused (no dA/dBx materialization), h[16] in registers.
// ---------------------------------------------------------------------------

#define D_MODEL 1024
#define D_STATE 16
#define D_CONV  4
#define D_INNER 2048
#define BATCH   2
#define SEQLEN  1024
#define NROWS   (BATCH * SEQLEN)          // 2048 rows of activations
#define XDBL_N  (D_INNER + 2 * D_STATE)   // 2080

typedef __attribute__((ext_vector_type(16))) __bf16      v16bf;
typedef __attribute__((ext_vector_type(8)))  float        v8f;
typedef __attribute__((ext_vector_type(8)))  unsigned int v8u;

// ---------------------------------------------------------------------------
// gfx1250 async global->LDS copy (ASYNCcnt-tracked, bypasses VGPRs).
// Inline asm: VDST = 32-bit LDS byte offset VGPR, VADDR = 64-bit global addr.
// ---------------------------------------------------------------------------
__device__ __forceinline__ unsigned lds_off_of(const void* p) {
  return (unsigned)(unsigned long long)(const __attribute__((address_space(3))) char*)p;
}

__device__ __forceinline__ void async_copy_b128(const __bf16* gsrc, __bf16* ldst) {
  const unsigned loff = lds_off_of(ldst);
  asm volatile("global_load_async_to_lds_b128 %0, %1, off"
               :: "v"(loff), "v"(gsrc)
               : "memory");
}

__device__ __forceinline__ void wait_async0() {
  asm volatile("s_wait_asynccnt 0x0" ::: "memory");
}

// ---------------------------------------------------------------------------
// LayerNorm -> bf16 (one block per row of 1024)
// ---------------------------------------------------------------------------
__global__ __launch_bounds__(256)
void mamba_ln_bf16(const float* __restrict__ x, const float* __restrict__ w,
                   const float* __restrict__ b, __bf16* __restrict__ out) {
  const int row = blockIdx.x;
  const float* xr = x + (size_t)row * D_MODEL;
  float v[4];
  float s = 0.f, s2 = 0.f;
#pragma unroll
  for (int i = 0; i < 4; ++i) {
    v[i] = xr[threadIdx.x + 256 * i];
    s += v[i]; s2 += v[i] * v[i];
  }
  __shared__ float rs[256], rq[256];
  rs[threadIdx.x] = s; rq[threadIdx.x] = s2;
  __syncthreads();
  for (int off = 128; off > 0; off >>= 1) {
    if (threadIdx.x < (unsigned)off) {
      rs[threadIdx.x] += rs[threadIdx.x + off];
      rq[threadIdx.x] += rq[threadIdx.x + off];
    }
    __syncthreads();
  }
  const float mu  = rs[0] * (1.f / D_MODEL);
  const float var = rq[0] * (1.f / D_MODEL) - mu * mu;
  const float inv = rsqrtf(var + 1e-5f);
#pragma unroll
  for (int i = 0; i < 4; ++i) {
    const int c = threadIdx.x + 256 * i;
    out[(size_t)row * D_MODEL + c] = (__bf16)((v[i] - mu) * inv * w[c] + b[c]);
  }
}

// ---------------------------------------------------------------------------
// Strided fp32 -> bf16 cast: dst[r*ncols+c] = src[r*src_stride + col0 + c]
// ---------------------------------------------------------------------------
__global__ __launch_bounds__(256)
void mamba_cast_bf16(const float* __restrict__ src, __bf16* __restrict__ dst,
                     int rows, int ncols, int src_stride, int col0) {
  const int i = blockIdx.x * 256 + threadIdx.x;
  if (i >= rows * ncols) return;
  const int r = i / ncols, c = i - r * ncols;
  dst[i] = (__bf16)src[(size_t)r * src_stride + col0 + c];
}

// ---------------------------------------------------------------------------
// WMMA fragment load from an LDS tile with row stride 40 bf16 elements.
// CDNA5 16-bit A-matrix 16x32 layout (B uses same pattern on W[n][k] = B^T):
//   lane m = lane&15, half = lane>>4
//   VGPR v<4 : K = half*8 + 2v ; VGPR v>=4 : K = 16 + half*8 + 2(v-4)
// ---------------------------------------------------------------------------
__device__ __forceinline__ v16bf ldsa_frag(const __bf16* base, int lane) {
  const int m  = lane & 15;
  const int hi = lane >> 4;
  v8u raw;
#pragma unroll
  for (int v = 0; v < 8; ++v) {
    const int k = ((v & 4) << 2) + hi * 8 + ((v & 3) << 1);
    raw[v] = *(const unsigned int*)(base + m * 40 + k);   // 2 bf16 per b32
  }
  return __builtin_bit_cast(v16bf, raw);
}

__device__ __forceinline__ void stage_tile(const __bf16* ga, const __bf16* gw,
                                           __bf16* la, __bf16* lw) {
  async_copy_b128(ga,     la);
  async_copy_b128(ga + 8, la + 8);
  async_copy_b128(gw,     lw);
  async_copy_b128(gw + 8, lw + 8);
}

// ---------------------------------------------------------------------------
// GEMM: C[M,N] = A[M,K](bf16) * W[N,K](bf16)^T   (f32 accumulate)
// epilogue: 0 = store, 1 = softplus(acc + bias[n]), 2 = acc + res[m*N+n]
// Tile 128x128, BK=32, 256 threads = 8 waves, each wave 32(M) x 64(N).
// Double-buffered LDS, async global->LDS staging. Requires M % 128 == 0 and
// K % 32 == 0 (true for every call here). N handled by row clamping: the
// clamped (duplicate) rows only feed output columns >= N, masked at store.
// ---------------------------------------------------------------------------
__global__ __launch_bounds__(256)
void mamba_gemm_bf16(const __bf16* __restrict__ A, const __bf16* __restrict__ W,
                     float* __restrict__ C, int M, int N, int K,
                     int epilogue, const float* __restrict__ bias,
                     const float* __restrict__ res) {
  __shared__ __align__(16) __bf16 As[2][128 * 40];
  __shared__ __align__(16) __bf16 Ws[2][128 * 40];

  const int tid  = threadIdx.x;
  const int lane = tid & 31;
  const int wid  = tid >> 5;
  const int wm   = (wid & 3) * 32;   // 4 waves tile M
  const int wn   = (wid >> 2) * 64;  // 2 waves tile N
  const int m0   = blockIdx.y * 128;
  const int n0   = blockIdx.x * 128;

  v8f acc[2][4];
#pragma unroll
  for (int i = 0; i < 2; ++i)
#pragma unroll
    for (int j = 0; j < 4; ++j) acc[i][j] = {};

  const int lrow = tid >> 1;          // 0..127
  const int lseg = (tid & 1) * 16;    // 0 or 16 (bf16 elements)
  const int arow = m0 + lrow;                                  // < M always
  const int wrow = (n0 + lrow < N) ? (n0 + lrow) : (N - 1);    // clamped
  const __bf16* gA = A + (size_t)arow * K + lseg;
  const __bf16* gW = W + (size_t)wrow * K + lseg;
  const unsigned sa = lrow * 40 + lseg;

  // prologue: stage tile 0 into buffer 0
  stage_tile(gA, gW, &As[0][sa], &Ws[0][sa]);
  wait_async0();
  __syncthreads();

  const int nk = K >> 5;
  for (int k = 0; k < nk; ++k) {
    const int cur = k & 1;
    if (k + 1 < nk)   // overlap: async-stage next tile into the other buffer
      stage_tile(gA + (size_t)(k + 1) * 32, gW + (size_t)(k + 1) * 32,
                 &As[cur ^ 1][sa], &Ws[cur ^ 1][sa]);

    v16bf af[2], bfg[4];
    af[0] = ldsa_frag(&As[cur][(wm + 0)  * 40], lane);
    af[1] = ldsa_frag(&As[cur][(wm + 16) * 40], lane);
#pragma unroll
    for (int j = 0; j < 4; ++j)
      bfg[j] = ldsa_frag(&Ws[cur][(wn + j * 16) * 40], lane);

#pragma unroll
    for (int i = 0; i < 2; ++i)
#pragma unroll
      for (int j = 0; j < 4; ++j)
        acc[i][j] = __builtin_amdgcn_wmma_f32_16x16x32_bf16(
            false, af[i], false, bfg[j], (short)0, acc[i][j], false, false);

    if (k + 1 < nk) {
      wait_async0();
      __syncthreads();
    }
  }

  // --- epilogue: C/D layout -> lane n = lane&15 ; VGPR r holds M = r + 8*(lane>>4)
  const int nloc = lane & 15;
  const int hi   = lane >> 4;
#pragma unroll
  for (int i = 0; i < 2; ++i)
#pragma unroll
    for (int j = 0; j < 4; ++j)
#pragma unroll
      for (int r = 0; r < 8; ++r) {
        const int gm = m0 + wm + i * 16 + r + 8 * hi;
        const int gn = n0 + wn + j * 16 + nloc;
        if (gn < N) {
          float v = acc[i][j][r];
          if (epilogue == 1) {                 // bias + softplus
            v += bias[gn];
            v = (v > 20.f) ? v : log1pf(__expf(v));
          } else if (epilogue == 2) {          // residual add
            v += res[(size_t)gm * N + gn];
          }
          C[(size_t)gm * N + gn] = v;
        }
      }
}

// ---------------------------------------------------------------------------
// Causal depthwise conv1d (k=4) + bias + SiLU over xi = zx[:, 2048:4096].
// Writes fp32 (for scan) and bf16 (for the x_dbl GEMM).
// ---------------------------------------------------------------------------
__global__ __launch_bounds__(256)
void mamba_conv_silu(const float* __restrict__ zx, const float* __restrict__ cw,
                     const float* __restrict__ cb, float* __restrict__ xi_f,
                     __bf16* __restrict__ xi_b) {
  const int i = blockIdx.x * 256 + threadIdx.x;   // over NROWS * D_INNER
  if (i >= NROWS * D_INNER) return;
  const int d  = i & (D_INNER - 1);
  const int bl = i >> 11;                         // b*L + l
  const int l  = bl & (SEQLEN - 1);
  const int b0 = bl - l;                          // b*L
  float acc = cb[d];
  const float* w = cw + d * D_CONV;
#pragma unroll
  for (int t = 0; t < D_CONV; ++t) {
    const int ls = l - (D_CONV - 1) + t;
    if (ls >= 0)
      acc += w[t] * zx[(size_t)(b0 + ls) * (2 * D_INNER) + D_INNER + d];
  }
  const float s = acc / (1.f + __expf(-acc));     // SiLU
  xi_f[i] = s;
  xi_b[i] = (__bf16)s;
}

// ---------------------------------------------------------------------------
// Selective scan, fused: h[16] in registers, exp(delta*A) computed on the fly.
// One thread per (b, d) channel; also fuses  y += xi * D.
// ---------------------------------------------------------------------------
__global__ __launch_bounds__(256)
void mamba_scan(const float* __restrict__ delta, const float* __restrict__ xi,
                const float* __restrict__ xdbl, const float* __restrict__ A_log,
                const float* __restrict__ Dv, float* __restrict__ y) {
  const int d = blockIdx.x * 256 + threadIdx.x;
  const int b = blockIdx.y;
  if (d >= D_INNER) return;
  float A[D_STATE], h[D_STATE];
#pragma unroll
  for (int n = 0; n < D_STATE; ++n) {
    A[n] = -__expf(A_log[d * D_STATE + n]);
    h[n] = 0.f;
  }
  const float Dd = Dv[d];
  for (int l = 0; l < SEQLEN; ++l) {
    const int r = b * SEQLEN + l;
    const float dt = delta[(size_t)r * D_INNER + d];
    const float xv = xi[(size_t)r * D_INNER + d];
    const float* Brow = xdbl + (size_t)r * XDBL_N + D_INNER;
    const float* Crow = Brow + D_STATE;
    const float dx = dt * xv;
    float acc = 0.f;
#pragma unroll
    for (int n = 0; n < D_STATE; ++n) {
      const float dA = __expf(dt * A[n]);
      h[n] = dA * h[n] + dx * Brow[n];
      acc += h[n] * Crow[n];
    }
    y[(size_t)r * D_INNER + d] = acc + xv * Dd;
  }
}

// ---------------------------------------------------------------------------
// Gating: y2 = y * silu(z), z = zx[:, 0:2048]; emit bf16 for the out GEMM.
// ---------------------------------------------------------------------------
__global__ __launch_bounds__(256)
void mamba_gate(const float* __restrict__ y, const float* __restrict__ zx,
                __bf16* __restrict__ y2) {
  const int i = blockIdx.x * 256 + threadIdx.x;
  if (i >= NROWS * D_INNER) return;
  const int d = i & (D_INNER - 1);
  const int r = i >> 11;
  const float z = zx[(size_t)r * (2 * D_INNER) + d];
  const float g = z / (1.f + __expf(-z));
  y2[i] = (__bf16)(y[i] * g);
}

// ---------------------------------------------------------------------------
// Host launcher
// ---------------------------------------------------------------------------
extern "C" void kernel_launch(void* const* d_in, const int* in_sizes, int n_in,
                              void* d_out, int out_size, void* d_ws, size_t ws_size,
                              hipStream_t stream) {
  const float* x      = (const float*)d_in[0];
  const float* W_in   = (const float*)d_in[1];
  const float* conv_w = (const float*)d_in[2];
  const float* conv_b = (const float*)d_in[3];
  const float* W_x    = (const float*)d_in[4];
  const float* W_dt   = (const float*)d_in[5];
  const float* b_dt   = (const float*)d_in[6];
  const float* A_log  = (const float*)d_in[7];
  const float* Dv     = (const float*)d_in[8];
  const float* W_out  = (const float*)d_in[9];
  const float* norm_w = (const float*)d_in[10];
  const float* norm_b = (const float*)d_in[11];
  float* out = (float*)d_out;

  char* ws = (char*)d_ws;
  const size_t SZ_A    = (size_t)NROWS * D_INNER * 2;          // 8 MB  (bf16 activations)
  const size_t SZ_W    = (size_t)XDBL_N * D_INNER * 2 + 256;   // 8.5 MB (bf16 weights, max)
  const size_t SZ_ZX   = (size_t)NROWS * 2 * D_INNER * 4;      // 32 MB
  const size_t SZ_XI   = (size_t)NROWS * D_INNER * 4;          // 16 MB
  const size_t SZ_XDBL = (size_t)NROWS * XDBL_N * 4;           // 17 MB
  const size_t SZ_DLT  = (size_t)NROWS * D_INNER * 4;          // 16 MB
  size_t off = 0;
  __bf16* bufA   = (__bf16*)(ws + off); off += SZ_A;
  __bf16* bufW   = (__bf16*)(ws + off); off += SZ_W;
  float*  zx     = (float*)(ws + off);  off += SZ_ZX;
  float*  xi_f   = (float*)(ws + off);  off += SZ_XI;
  float*  xdbl   = (float*)(ws + off);  off += SZ_XDBL;
  float*  dlt    = (float*)(ws + off);  off += SZ_DLT;
  float*  y_f    = (float*)(ws + off);  off += SZ_XI;
  (void)ws_size; (void)in_sizes; (void)n_in; (void)out_size;

  const int ew = 256;
  const int nElem = NROWS * D_INNER;

  // 1) LayerNorm -> bf16 activations
  mamba_ln_bf16<<<NROWS, ew, 0, stream>>>(x, norm_w, norm_b, bufA);

  // 2) zx = xn @ W_in^T   [2048 x 4096], K=1024
  mamba_cast_bf16<<<(2 * D_INNER * D_MODEL + ew - 1) / ew, ew, 0, stream>>>(
      W_in, bufW, 2 * D_INNER, D_MODEL, D_MODEL, 0);
  mamba_gemm_bf16<<<dim3((2 * D_INNER) / 128, NROWS / 128), ew, 0, stream>>>(
      bufA, bufW, zx, NROWS, 2 * D_INNER, D_MODEL, 0, nullptr, nullptr);

  // 3) causal depthwise conv + SiLU -> xi (fp32 + bf16)
  mamba_conv_silu<<<(nElem + ew - 1) / ew, ew, 0, stream>>>(
      zx, conv_w, conv_b, xi_f, bufA);

  // 4) x_dbl = xi @ W_x^T  [2048 x 2080], K=2048
  mamba_cast_bf16<<<(XDBL_N * D_INNER + ew - 1) / ew, ew, 0, stream>>>(
      W_x, bufW, XDBL_N, D_INNER, D_INNER, 0);
  mamba_gemm_bf16<<<dim3((XDBL_N + 127) / 128, NROWS / 128), ew, 0, stream>>>(
      bufA, bufW, xdbl, NROWS, XDBL_N, D_INNER, 0, nullptr, nullptr);

  // 5) delta = softplus(x_dbl[:, :2048] @ W_dt^T + b_dt)  [2048 x 2048], K=2048
  mamba_cast_bf16<<<(nElem + ew - 1) / ew, ew, 0, stream>>>(
      xdbl, bufA, NROWS, D_INNER, XDBL_N, 0);
  mamba_cast_bf16<<<(D_INNER * D_INNER + ew - 1) / ew, ew, 0, stream>>>(
      W_dt, bufW, D_INNER, D_INNER, D_INNER, 0);
  mamba_gemm_bf16<<<dim3(D_INNER / 128, NROWS / 128), ew, 0, stream>>>(
      bufA, bufW, dlt, NROWS, D_INNER, D_INNER, 1, b_dt, nullptr);

  // 6) selective scan (fused exp, + xi*D)
  mamba_scan<<<dim3(D_INNER / ew, BATCH), ew, 0, stream>>>(
      dlt, xi_f, xdbl, A_log, Dv, y_f);

  // 7) gating y2 = y * silu(z) -> bf16
  mamba_gate<<<(nElem + ew - 1) / ew, ew, 0, stream>>>(y_f, zx, bufA);

  // 8) out = y2 @ W_out^T + residual  [2048 x 1024], K=2048
  mamba_cast_bf16<<<(D_MODEL * D_INNER + ew - 1) / ew, ew, 0, stream>>>(
      W_out, bufW, D_MODEL, D_INNER, D_INNER, 0);
  mamba_gemm_bf16<<<dim3(D_MODEL / 128, NROWS / 128), ew, 0, stream>>>(
      bufA, bufW, out, NROWS, D_MODEL, D_INNER, 2, nullptr, x);
}
